// ConcreteDropout_lstm_80994493268407
// MI455X (gfx1250) — matrix operand order repaired and checked by reading, not compiled, not run on hardware
//
#include <hip/hip_runtime.h>
#include <hip/hip_bf16.h>
#include <cstdint>
#include <cstddef>

// ---------------------------------------------------------------------------
// Problem constants (from reference): B=256, T=512, D=128, H=128, G=4H=512
// ---------------------------------------------------------------------------
#define BB   256
#define TT   512
#define DD   128
#define HH   128
#define GG   512
#define GLD  520   // padded gate row stride (520 mod 64 banks = 8 -> no conflicts)

#define EPSF  1e-7f
#define TEMPF 0.1f

typedef __attribute__((ext_vector_type(16))) _Float16 v16h;
typedef __attribute__((ext_vector_type(8)))  float    v8f;

union FragU { v16h v; uint4 q[2]; };

__device__ __forceinline__ float sigf(float x) {
    return 1.0f / (1.0f + __expf(-x));
}
// robust tanh: 1 - 2/(e^{2x}+1); correct limits at +-inf, no NaN
__device__ __forceinline__ float tanhfast(float x) {
    return 1.0f - 2.0f / (__expf(2.0f * x) + 1.0f);
}

// B-fragment (K x N = 32 x 16, f16) for v_wmma_f32_16x16x32_f16.
// K order chosen here (not the ISA table's): lane holds column n = lane&15; lanes 0-15 hold K=0..15,
// lanes 16-31 hold K=16..31, 2 values per VGPR -> 16 *contiguous* K values
// per lane when W is stored row-major [N][K]. 32 contiguous bytes per lane.
__device__ __forceinline__ v16h load_bfrag(const _Float16* __restrict__ W,
                                           int n0, int lane, int kk) {
    int n    = n0 + (lane & 15);
    int koff = kk * 32 + (lane >> 4) * 16;
    const uint4* p = reinterpret_cast<const uint4*>(W + n * DD + koff);
    FragU u; u.q[0] = p[0]; u.q[1] = p[1];
    return u.v;
}

// A-fragment (M x K = 16 x 32, f16) from an LDS 16x128 row-major tile.
// lane m = lane&15; lanes 0-15 hold K=0..15, lanes 16-31 hold K=16..31
// (contiguous, matching load_bfrag) -> two contiguous 16B chunks per lane.
__device__ __forceinline__ v16h load_afrag(const _Float16* __restrict__ A,
                                           int lane, int kk) {
    int m    = lane & 15;
    int half = lane >> 4;
    const _Float16* base = A + m * DD + kk * 32;
    FragU u;
    u.q[0] = *reinterpret_cast<const uint4*>(base + half * 16);
    u.q[1] = *reinterpret_cast<const uint4*>(base + half * 16 + 8);
    return u.v;
}

// ---------------------------------------------------------------------------
// Kernel 1: scalars + regularization (single block reduction)
// outreg[0]=regularization, [1]=p, [2]=p_rec
// ---------------------------------------------------------------------------
__global__ void prep_reg_kernel(const float* __restrict__ wih,
                                const float* __restrict__ whh,
                                const float* __restrict__ bih,
                                const float* __restrict__ bhh,
                                const float* __restrict__ plog,
                                const float* __restrict__ plogr,
                                float* __restrict__ outreg) {
    __shared__ float red0[256], red1[256], red2[256];
    int tid = threadIdx.x;
    float s0 = 0.f, s1 = 0.f, s2 = 0.f;
    for (int i = tid; i < GG * DD; i += 256) {
        float a = wih[i]; s0 += a * a;
        float b = whh[i]; s1 += b * b;
    }
    for (int i = tid; i < GG; i += 256) {
        float a = bih[i]; s2 += a * a;
        float b = bhh[i]; s2 += b * b;
    }
    red0[tid] = s0; red1[tid] = s1; red2[tid] = s2;
    __syncthreads();
    for (int s = 128; s > 0; s >>= 1) {
        if (tid < s) {
            red0[tid] += red0[tid + s];
            red1[tid] += red1[tid + s];
            red2[tid] += red2[tid + s];
        }
        __syncthreads();
    }
    if (tid == 0) {
        float p  = sigf(plog[0]);
        float pr = sigf(plogr[0]);
        float wreg = 1e-6f * (red0[0] / (1.f - p) + red1[0] / (1.f - pr));
        float breg = 1e-6f * red2[0];
        float entk = p * __logf(p) + (1.f - p) * __logf(1.f - p);
        float entr = pr * __logf(pr) + (1.f - pr) * __logf(1.f - pr);
        float dreg = 1e-5f * ((float)DD * entk + (float)HH * entr);
        outreg[0] = wreg + breg + dreg;
        outreg[1] = p;
        outreg[2] = pr;
    }
}

// ---------------------------------------------------------------------------
// Kernel 2: concrete-dropout masks (pre-scaled by 1/(1-p)), combined bias,
// and f32 -> f16 weight conversion into workspace.
// ---------------------------------------------------------------------------
__global__ void prep_masks_kernel(const float* __restrict__ wih,
                                  const float* __restrict__ whh,
                                  const float* __restrict__ bih,
                                  const float* __restrict__ bhh,
                                  const float* __restrict__ plog,
                                  const float* __restrict__ plogr,
                                  const float* __restrict__ ux,
                                  const float* __restrict__ uh,
                                  float* __restrict__ mx,
                                  float* __restrict__ mh,
                                  float* __restrict__ bs,
                                  _Float16* __restrict__ wih_h,
                                  _Float16* __restrict__ whh_h) {
    int i = blockIdx.x * blockDim.x + threadIdx.x;
    float p  = sigf(plog[0]);
    float pr = sigf(plogr[0]);
    if (i < GG * DD) {
        wih_h[i] = (_Float16)wih[i];
        whh_h[i] = (_Float16)whh[i];
    }
    if (i < GG) bs[i] = bih[i] + bhh[i];
    if (i < BB * DD) {
        float pl  = __logf(p + EPSF)  - __logf(1.f - p + EPSF);
        float plr = __logf(pr + EPSF) - __logf(1.f - pr + EPSF);
        float u   = ux[i];
        float lg  = pl + __logf(u + EPSF) - __logf(1.f - u + EPSF);
        mx[i] = (1.f - sigf(lg / TEMPF)) / (1.f - p);
        float u2  = uh[i];
        float lg2 = plr + __logf(u2 + EPSF) - __logf(1.f - u2 + EPSF);
        mh[i] = (1.f - sigf(lg2 / TEMPF)) / (1.f - pr);
    }
}

// ---------------------------------------------------------------------------
// Kernel 3: persistent-state LSTM. 16 blocks x 512 threads (16 waves).
// Each block owns 16 batch rows; h/c live in LDS/registers for all 512 steps.
// Weights live as f16 B-fragments in registers for the whole kernel.
// Per step, each wave: hoist 8 A-frag LDS loads, then 16 WMMAs across
// 4 independent accumulation chains (2 N-tiles x even/odd K) for pipelining.
// ---------------------------------------------------------------------------
__global__ void __launch_bounds__(512, 1)
lstm_kernel(const float* __restrict__ xg,
            const float* __restrict__ mx,
            const float* __restrict__ mh,
            const float* __restrict__ bs,
            const _Float16* __restrict__ wih_h,
            const _Float16* __restrict__ whh_h,
            float* __restrict__ xout,
            float* __restrict__ hlast) {
    __shared__ _Float16 xbf[16 * DD];   // masked xt, f16 (WMMA A source)
    __shared__ _Float16 hbf[16 * DD];   // masked h,  f16 (WMMA A source)
    __shared__ float    gates[16 * GLD];
    __shared__ float    mxL[16 * DD];
    __shared__ float    mhL[16 * DD];
    __shared__ float    bsL[GG];

    const int tid  = threadIdx.x;
    const int lane = tid & 31;
    const int wave = tid >> 5;          // 0..15
    const int b0   = blockIdx.x * 16;   // first batch row of this block

    // ---- init: masks/bias to LDS, h=0, c=0 --------------------------------
    for (int i = tid; i < 16 * DD; i += 512) {
        mxL[i] = mx[(size_t)b0 * DD + (i >> 7) * DD + (i & 127)];
        mhL[i] = mh[(size_t)b0 * HH + (i >> 7) * HH + (i & 127)];
        hbf[i] = (_Float16)0.0f;
    }
    for (int i = tid; i < GG; i += 512) bsL[i] = bs[i];

    // ---- preload B-fragments (weights) into registers ---------------------
    const int n0a = wave * 32;          // this wave's two 16-wide N tiles
    const int n0b = wave * 32 + 16;
    v16h Bi0[4], Bi1[4], Bh0[4], Bh1[4];
#pragma unroll
    for (int kk = 0; kk < 4; ++kk) {
        Bi0[kk] = load_bfrag(wih_h, n0a, lane, kk);
        Bi1[kk] = load_bfrag(wih_h, n0b, lane, kk);
        Bh0[kk] = load_bfrag(whh_h, n0a, lane, kk);
        Bh1[kk] = load_bfrag(whh_h, n0b, lane, kk);
    }

    float creg[4] = {0.f, 0.f, 0.f, 0.f};
    __syncthreads();

    // x staging indices: one float4 per thread (16 rows * 32 float4 = 512)
    const int sm = tid >> 5;            // row 0..15
    const int sd = (tid & 31) * 4;      // col 0..124
    const float* xrow = xg + ((size_t)(b0 + sm) * TT) * DD + sd;

    for (int t = 0; t < TT; ++t) {
        // ---- stage masked xt into LDS as f16 ------------------------------
        {
            float4 xr = *reinterpret_cast<const float4*>(xrow + (size_t)t * DD);
            union { _Float16 h[4]; uint2 u; } hv;
            hv.h[0] = (_Float16)(xr.x * mxL[sm * DD + sd + 0]);
            hv.h[1] = (_Float16)(xr.y * mxL[sm * DD + sd + 1]);
            hv.h[2] = (_Float16)(xr.z * mxL[sm * DD + sd + 2]);
            hv.h[3] = (_Float16)(xr.w * mxL[sm * DD + sd + 3]);
            *reinterpret_cast<uint2*>(&xbf[sm * DD + sd]) = hv.u;
            if (t + 1 < TT)  // CDNA5 global_prefetch_b8 for next timestep
                __builtin_prefetch(xrow + (size_t)(t + 1) * DD, 0, 1);
        }
        __syncthreads();    // xbf + hbf ready

        // ---- hoist all 8 A-fragments (16 ds_load_b128 pipelined) ----------
        v16h ax[4], ah[4];
#pragma unroll
        for (int kk = 0; kk < 4; ++kk) ax[kk] = load_afrag(xbf, lane, kk);
#pragma unroll
        for (int kk = 0; kk < 4; ++kk) ah[kk] = load_afrag(hbf, lane, kk);

        // ---- 16 back-to-back WMMAs over 4 independent chains --------------
        v8f p0 = {}, p1 = {}, q0 = {}, q1 = {};
#pragma unroll
        for (int kk = 0; kk < 4; kk += 2) {
            p0 = __builtin_amdgcn_wmma_f32_16x16x32_f16(
                false, ax[kk],     false, Bi0[kk],     (short)0, p0, false, false);
            q0 = __builtin_amdgcn_wmma_f32_16x16x32_f16(
                false, ax[kk + 1], false, Bi0[kk + 1], (short)0, q0, false, false);
            p1 = __builtin_amdgcn_wmma_f32_16x16x32_f16(
                false, ax[kk],     false, Bi1[kk],     (short)0, p1, false, false);
            q1 = __builtin_amdgcn_wmma_f32_16x16x32_f16(
                false, ax[kk + 1], false, Bi1[kk + 1], (short)0, q1, false, false);
        }
#pragma unroll
        for (int kk = 0; kk < 4; kk += 2) {
            p0 = __builtin_amdgcn_wmma_f32_16x16x32_f16(
                false, ah[kk],     false, Bh0[kk],     (short)0, p0, false, false);
            q0 = __builtin_amdgcn_wmma_f32_16x16x32_f16(
                false, ah[kk + 1], false, Bh0[kk + 1], (short)0, q0, false, false);
            p1 = __builtin_amdgcn_wmma_f32_16x16x32_f16(
                false, ah[kk],     false, Bh1[kk],     (short)0, p1, false, false);
            q1 = __builtin_amdgcn_wmma_f32_16x16x32_f16(
                false, ah[kk + 1], false, Bh1[kk + 1], (short)0, q1, false, false);
        }
        v8f acc0 = p0 + q0;
        v8f acc1 = p1 + q1;

        // C/D layout: VGPR r -> row M = (lane>>4)*8 + r, col N = lane&15
#pragma unroll
        for (int r = 0; r < 8; ++r) {
            int mrow = (lane >> 4) * 8 + r;
            gates[mrow * GLD + n0a + (lane & 15)] = acc0[r];
            gates[mrow * GLD + n0b + (lane & 15)] = acc1[r];
        }
        __syncthreads();    // gates ready

        // ---- elementwise LSTM update (torch gate order: i, f, g, o) -------
#pragma unroll
        for (int k = 0; k < 4; ++k) {
            int idx = tid + k * 512;      // 0..2047
            int m   = idx >> 7;           // row 0..15
            int hh  = idx & 127;          // col 0..127
            float ig = gates[m * GLD + hh]             + bsL[hh];
            float fg = gates[m * GLD + 128 + hh]       + bsL[128 + hh];
            float gg = gates[m * GLD + 256 + hh]       + bsL[256 + hh];
            float og = gates[m * GLD + 384 + hh]       + bsL[384 + hh];
            float c  = sigf(fg) * creg[k] + sigf(ig) * tanhfast(gg);
            float h  = sigf(og) * tanhfast(c);
            creg[k]  = c;
            xout[((size_t)(b0 + m) * TT + t) * HH + hh] = h;
            hbf[m * DD + hh] = (_Float16)(h * mhL[m * DD + hh]);
            if (t == TT - 1) hlast[(size_t)(b0 + m) * HH + hh] = h;
        }
        // next iteration's staging writes xbf only; hbf/gates hazards are
        // covered by the two barriers above.
    }
}

// ---------------------------------------------------------------------------
// Host launcher
// ---------------------------------------------------------------------------
extern "C" void kernel_launch(void* const* d_in, const int* in_sizes, int n_in,
                              void* d_out, int out_size, void* d_ws, size_t ws_size,
                              hipStream_t stream) {
    (void)in_sizes; (void)n_in; (void)out_size; (void)ws_size;
    const float* x     = (const float*)d_in[0];
    const float* wih   = (const float*)d_in[1];
    const float* whh   = (const float*)d_in[2];
    const float* bih   = (const float*)d_in[3];
    const float* bhh   = (const float*)d_in[4];
    const float* plog  = (const float*)d_in[5];
    const float* plogr = (const float*)d_in[6];
    const float* ux    = (const float*)d_in[7];
    const float* uh    = (const float*)d_in[8];

    float* out = (float*)d_out;
    // output layout: x_out [B,T,H] | h_last [B,H] | reg | p | p_rec
    const size_t XOUT_N  = (size_t)BB * TT * HH;   // 16777216
    const size_t HLAST_N = (size_t)BB * HH;        // 32768
    float* xout  = out;
    float* hlast = out + XOUT_N;
    float* regs  = out + XOUT_N + HLAST_N;

    // workspace layout (bytes)
    char* ws = (char*)d_ws;
    float*    mx    = (float*)(ws + 0);                 // 32768 f32
    float*    mh    = (float*)(ws + 131072);            // 32768 f32
    float*    bs    = (float*)(ws + 262144);            //   512 f32
    _Float16* wih_h = (_Float16*)(ws + 264192);         // 65536 f16
    _Float16* whh_h = (_Float16*)(ws + 395264);         // 65536 f16

    prep_reg_kernel<<<1, 256, 0, stream>>>(wih, whh, bih, bhh, plog, plogr, regs);
    prep_masks_kernel<<<(GG * DD + 255) / 256, 256, 0, stream>>>(
        wih, whh, bih, bhh, plog, plogr, ux, uh, mx, mh, bs, wih_h, whh_h);
    lstm_kernel<<<16, 512, 0, stream>>>(x, mx, mh, bs, wih_h, whh_h, xout, hlast);
}
